// LINnet_D_43920335569292
// MI455X (gfx1250) — matrix operand order repaired and checked by reading, not compiled, not run on hardware
//
#include <hip/hip_runtime.h>

typedef __attribute__((ext_vector_type(16))) _Float16 v16h;
typedef __attribute__((ext_vector_type(8)))  _Float16 v8h;
typedef __attribute__((ext_vector_type(8)))  float    v8f;

#define BATCH 512
#define BN_EPS 1e-5f
#define SLOPE 0.2f

// ---------------------------------------------------------------------------
// Pad W1 [64,1,4,4] fp32 -> f16 [64][32] with K 16..31 zeroed (K-pad for WMMA).
// ---------------------------------------------------------------------------
__global__ void w1_pad_kernel(const float* __restrict__ W1, _Float16* __restrict__ out) {
    int i = blockIdx.x * blockDim.x + threadIdx.x;
    if (i >= 64 * 32) return;
    int oc = i >> 5, r = i & 31;
    out[i] = (r < 16) ? (_Float16)W1[oc * 16 + r] : (_Float16)0.f;
}

__global__ void f32_to_f16_kernel(const float* __restrict__ in,
                                  _Float16* __restrict__ out, int n) {
    int i = blockIdx.x * blockDim.x + threadIdx.x;
    if (i < n) out[i] = (_Float16)in[i];
}

// ---------------------------------------------------------------------------
// im2col of the raw input x [512,1,48,80] -> col [N=512*24*40][32] f16,
// K index r: r<16 -> 4x4 patch (zero-padded borders), r>=16 -> 0 (K padding).
// ---------------------------------------------------------------------------
__global__ void im2col_x_kernel(const float* __restrict__ x, _Float16* __restrict__ col) {
    const int total = BATCH * 24 * 40 * 32;
    int idx = blockIdx.x * blockDim.x + threadIdx.x;
    if (idx >= total) return;
    int n = idx >> 5, r = idx & 31;
    _Float16 v = (_Float16)0.f;
    if (r < 16) {
        int ky = r >> 2, kx = r & 3;
        int ox = n % 40;
        int oy = (n / 40) % 24;
        int b  = n / (40 * 24);
        int iy = oy * 2 - 1 + ky, ix = ox * 2 - 1 + kx;
        if (iy >= 0 && iy < 48 && ix >= 0 && ix < 80)
            v = (_Float16)x[(size_t)b * 48 * 80 + iy * 80 + ix];
    }
    col[idx] = v;
}

// ---------------------------------------------------------------------------
// Per-channel batch statistics over (N,H,W): one block per channel.
// ---------------------------------------------------------------------------
__global__ void stats_kernel(const float* __restrict__ h, int C, int HW,
                             float* __restrict__ mean, float* __restrict__ istd) {
    int c = blockIdx.x;
    int n = BATCH * HW;
    float s = 0.f, s2 = 0.f;
    for (int i = threadIdx.x; i < n; i += blockDim.x) {
        int b = i / HW;
        int p = i - b * HW;
        float v = h[((size_t)b * C + c) * HW + p];
        s += v; s2 += v * v;
    }
    __shared__ float sh[256], sh2[256];
    sh[threadIdx.x] = s; sh2[threadIdx.x] = s2;
    __syncthreads();
    for (int st = 128; st > 0; st >>= 1) {
        if ((int)threadIdx.x < st) {
            sh[threadIdx.x]  += sh[threadIdx.x + st];
            sh2[threadIdx.x] += sh2[threadIdx.x + st];
        }
        __syncthreads();
    }
    if (threadIdx.x == 0) {
        float m   = sh[0] / (float)n;
        float var = sh2[0] / (float)n - m * m;
        mean[c] = m;
        istd[c] = rsqrtf(var + BN_EPS);
    }
}

// ---------------------------------------------------------------------------
// Fused BN(train) + LeakyReLU + im2col: raw [B,IC,IH,IW] fp32 ->
// col [B*OH*OW][IC*16] f16 for the next stride-2 4x4 conv (zero-padded borders).
// ---------------------------------------------------------------------------
template <int IC, int IH, int IW, int OH, int OW>
__global__ void norm_im2col_kernel(const float* __restrict__ raw,
                                   const float* __restrict__ mean,
                                   const float* __restrict__ istd,
                                   const float* __restrict__ gamma,
                                   const float* __restrict__ beta,
                                   _Float16* __restrict__ col) {
    constexpr int K = IC * 16;
    const int total = BATCH * OH * OW * K;
    for (int idx = blockIdx.x * blockDim.x + threadIdx.x; idx < total;
         idx += gridDim.x * blockDim.x) {
        int n = idx / K;          // K is a power of two -> shift
        int r = idx & (K - 1);
        int ic = r >> 4;
        int ky = (r >> 2) & 3, kx = r & 3;
        int ox = n % OW;
        int t  = n / OW;
        int oy = t % OH;
        int b  = t / OH;
        int iy = oy * 2 - 1 + ky, ix = ox * 2 - 1 + kx;
        _Float16 v = (_Float16)0.f;
        if (iy >= 0 && iy < IH && ix >= 0 && ix < IW) {
            float u = raw[(((size_t)b * IC + ic) * IH + iy) * IW + ix];
            u = (u - mean[ic]) * istd[ic] * gamma[ic] + beta[ic];
            u = (u >= 0.f) ? u : SLOPE * u;
            v = (_Float16)u;
        }
        col[idx] = v;
    }
}

// ---------------------------------------------------------------------------
// BN + LeakyReLU only (fp32 out) for the final layer feeding the head.
// ---------------------------------------------------------------------------
__global__ void norm_lrelu_f32_kernel(const float* __restrict__ h,
                                      const float* __restrict__ mean,
                                      const float* __restrict__ istd,
                                      const float* __restrict__ gamma,
                                      const float* __restrict__ beta,
                                      float* __restrict__ out, int C, int HW, int total) {
    for (int i = blockIdx.x * blockDim.x + threadIdx.x; i < total;
         i += gridDim.x * blockDim.x) {
        int c = (i / HW) % C;
        float v = (h[i] - mean[c]) * istd[c] * gamma[c] + beta[c];
        out[i] = (v >= 0.f) ? v : SLOPE * v;
    }
}

// ---------------------------------------------------------------------------
// WMMA GEMM: out[oc,n] = sum_K wgt[oc][K] * col[n][K]   (both K-contiguous f16)
// One wave computes a 64(M) x 32(N) fp32 tile: 8 accumulators, A fragments
// reused across both N halves -> 8 v_wmma per K=32 step vs 12 b128 loads.
// Fragment layouts per CDNA5 ISA 7.12.2 (wave32).
// ---------------------------------------------------------------------------
template <int K, int OC, int OH, int OW>
__global__ void conv_wmma_kernel(const _Float16* __restrict__ col, // [N][K]
                                 const _Float16* __restrict__ wgt, // [OC][K]
                                 float* __restrict__ out) {        // [B,OC,OH,OW]
    constexpr int MG = OC / 64;
    constexpr int KT = K / 32;
    const int wid  = blockIdx.x * (blockDim.x >> 5) + (threadIdx.x >> 5);
    const int lane = threadIdx.x & 31;
    const int half = lane >> 4;
    const int lm   = lane & 15;

    const int mg = wid % MG;
    const int nt = wid / MG;

    int nn[2];
    const _Float16* bptr[2];
#pragma unroll
    for (int nb = 0; nb < 2; ++nb) {
        nn[nb] = nt * 32 + nb * 16 + lm;
        // B fragment: lane holds K = kt*32 + half*16 + i, i=0..15 contiguous
        bptr[nb] = col + (size_t)nn[nb] * K + half * 16;
    }
    const _Float16* aptr[4];
#pragma unroll
    for (int t = 0; t < 4; ++t)
        // A fragment: lane holds K = kt*32 + half*16 + i, i=0..15 contiguous
        aptr[t] = wgt + (size_t)(mg * 64 + t * 16 + lm) * K + half * 16;

    v8f acc[4][2] = {};
    for (int kt = 0; kt < KT; ++kt) {
        // stream ahead (global_prefetch_b8)
        if (kt + 4 < KT) {
            __builtin_prefetch((const void*)(bptr[0] + (kt + 4) * 32), 0, 0);
            __builtin_prefetch((const void*)(bptr[1] + (kt + 4) * 32), 0, 0);
        }
        v16h bfrag[2];
#pragma unroll
        for (int nb = 0; nb < 2; ++nb) {
            v8h lo = *(const v8h*)(bptr[nb] + kt * 32);
            v8h hi = *(const v8h*)(bptr[nb] + kt * 32 + 8);
#pragma unroll
            for (int i = 0; i < 8; ++i) { bfrag[nb][i] = lo[i]; bfrag[nb][8 + i] = hi[i]; }
        }
#pragma unroll
        for (int t = 0; t < 4; ++t) {
            v8h lo = *(const v8h*)(aptr[t] + kt * 32);
            v8h hi = *(const v8h*)(aptr[t] + kt * 32 + 8);
            v16h a;
#pragma unroll
            for (int i = 0; i < 8; ++i) { a[i] = lo[i]; a[8 + i] = hi[i]; }
#pragma unroll
            for (int nb = 0; nb < 2; ++nb)
                acc[t][nb] = __builtin_amdgcn_wmma_f32_16x16x32_f16(
                    false, a, false, bfrag[nb], (short)0, acc[t][nb], false, false);
        }
    }

    // D layout: VGPR r -> M = r + 8*half, N = lm
#pragma unroll
    for (int nb = 0; nb < 2; ++nb) {
        int n  = nn[nb];
        int ox = n % OW;
        int oy = (n / OW) % OH;
        int b  = n / (OW * OH);
#pragma unroll
        for (int t = 0; t < 4; ++t) {
#pragma unroll
            for (int r = 0; r < 8; ++r) {
                int oc = mg * 64 + t * 16 + r + 8 * half;
                out[(((size_t)b * OC + oc) * OH + oy) * OW + ox] = acc[t][nb][r];
            }
        }
    }
}

// ---------------------------------------------------------------------------
// Label-routed head: out[b] = dot(h4n[b], W5[y[b]]) over 512*3*5 elements.
// ---------------------------------------------------------------------------
__global__ void head_kernel(const float* __restrict__ h4,  // [512,512,3,5]
                            const float* __restrict__ W5,  // [41,512,3,5]
                            const int* __restrict__ y,
                            float* __restrict__ out) {
    int b = blockIdx.x;
    int e = y[b];
    const float* hb = h4 + (size_t)b * 512 * 15;
    const float* wb = W5 + (size_t)e * 512 * 15;
    float s = 0.f;
    for (int i = threadIdx.x; i < 512 * 15; i += blockDim.x) s += hb[i] * wb[i];
    __shared__ float sh[256];
    sh[threadIdx.x] = s;
    __syncthreads();
    for (int st = 128; st > 0; st >>= 1) {
        if ((int)threadIdx.x < st) sh[threadIdx.x] += sh[threadIdx.x + st];
        __syncthreads();
    }
    if (threadIdx.x == 0) out[b] = sh[0];
}

// ---------------------------------------------------------------------------
extern "C" void kernel_launch(void* const* d_in, const int* in_sizes, int n_in,
                              void* d_out, int out_size, void* d_ws, size_t ws_size,
                              hipStream_t stream) {
    (void)in_sizes; (void)n_in; (void)out_size; (void)ws_size;

    const float* x  = (const float*)d_in[0];
    const float* W1 = (const float*)d_in[1];
    const float* g1 = (const float*)d_in[2];
    const float* b1 = (const float*)d_in[3];
    const float* W2 = (const float*)d_in[4];
    const float* g2 = (const float*)d_in[5];
    const float* b2 = (const float*)d_in[6];
    const float* W3 = (const float*)d_in[7];
    const float* g3 = (const float*)d_in[8];
    const float* b3 = (const float*)d_in[9];
    const float* W4 = (const float*)d_in[10];
    const float* g4 = (const float*)d_in[11];
    const float* b4 = (const float*)d_in[12];
    const float* W5 = (const float*)d_in[13];
    const int*   y  = (const int*)d_in[14];
    float* out = (float*)d_out;

    // --- workspace: one raw region + one col region, reused across layers ---
    constexpr size_t RAWMAX = (size_t)512 * 64 * 24 * 40;      // fp32 elements (layer1 max)
    constexpr size_t COLMAX = (size_t)512 * 12 * 20 * 1024;    // f16 halves   (col1 max)
    constexpr size_t H4 = (size_t)512 * 512 * 3 * 5;
    char* ws = (char*)d_ws;
    size_t o = 0;
    float*    raw = (float*)(ws + o);    o += RAWMAX * 4;      // 125.8 MB
    _Float16* col = (_Float16*)(ws + o); o += COLMAX * 2;      // 251.7 MB
    float*    h4n = (float*)(ws + o);    o += H4 * 4;          //  15.7 MB
    _Float16* w1h = (_Float16*)(ws + o); o += (size_t)64 * 32 * 2;
    _Float16* w2h = (_Float16*)(ws + o); o += (size_t)128 * 1024 * 2;
    _Float16* w3h = (_Float16*)(ws + o); o += (size_t)256 * 2048 * 2;
    _Float16* w4h = (_Float16*)(ws + o); o += (size_t)512 * 4096 * 2;
    float* mean1 = (float*)(ws + o); o += 64 * 4;
    float* istd1 = (float*)(ws + o); o += 64 * 4;
    float* mean2 = (float*)(ws + o); o += 128 * 4;
    float* istd2 = (float*)(ws + o); o += 128 * 4;
    float* mean3 = (float*)(ws + o); o += 256 * 4;
    float* istd3 = (float*)(ws + o); o += 256 * 4;
    float* mean4 = (float*)(ws + o); o += 512 * 4;
    float* istd4 = (float*)(ws + o); o += 512 * 4;

    // --- weight conversions (independent) ---
    w1_pad_kernel<<<(64 * 32 + 255) / 256, 256, 0, stream>>>(W1, w1h);
    f32_to_f16_kernel<<<(128 * 1024 + 255) / 256, 256, 0, stream>>>(W2, w2h, 128 * 1024);
    f32_to_f16_kernel<<<(256 * 2048 + 255) / 256, 256, 0, stream>>>(W3, w3h, 256 * 2048);
    f32_to_f16_kernel<<<(512 * 4096 + 255) / 256, 256, 0, stream>>>(W4, w4h, 512 * 4096);

    // --- layer 1: im2col(x) -> WMMA (K padded to 32) ---
    im2col_x_kernel<<<(BATCH * 24 * 40 * 32 + 255) / 256, 256, 0, stream>>>(x, col);
    // waves = 1 * (512*24*40/32) = 15360 -> 3840 blocks of 4 waves
    conv_wmma_kernel<32, 64, 24, 40><<<3840, 128, 0, stream>>>(col, w1h, raw);
    stats_kernel<<<64, 256, 0, stream>>>(raw, 64, 24 * 40, mean1, istd1);
    norm_im2col_kernel<64, 24, 40, 12, 20><<<8192, 256, 0, stream>>>(
        raw, mean1, istd1, g1, b1, col);

    // --- layer 2: waves = 2 * (512*12*20/32) = 7680 -> 1920 blocks ---
    conv_wmma_kernel<1024, 128, 12, 20><<<1920, 128, 0, stream>>>(col, w2h, raw);
    stats_kernel<<<128, 256, 0, stream>>>(raw, 128, 240, mean2, istd2);
    norm_im2col_kernel<128, 12, 20, 6, 10><<<8192, 256, 0, stream>>>(
        raw, mean2, istd2, g2, b2, col);

    // --- layer 3: waves = 4 * (512*6*10/32) = 3840 -> 960 blocks ---
    conv_wmma_kernel<2048, 256, 6, 10><<<960, 128, 0, stream>>>(col, w3h, raw);
    stats_kernel<<<256, 256, 0, stream>>>(raw, 256, 60, mean3, istd3);
    norm_im2col_kernel<256, 6, 10, 3, 5><<<8192, 256, 0, stream>>>(
        raw, mean3, istd3, g3, b3, col);

    // --- layer 4: waves = 8 * (512*3*5/32) = 1920 -> 480 blocks ---
    conv_wmma_kernel<4096, 512, 3, 5><<<480, 128, 0, stream>>>(col, w4h, raw);
    stats_kernel<<<512, 256, 0, stream>>>(raw, 512, 15, mean4, istd4);
    norm_lrelu_f32_kernel<<<2048, 256, 0, stream>>>(raw, mean4, istd4, g4, b4,
                                                    h4n, 512, 15, (int)H4);

    // --- routed head ---
    head_kernel<<<512, 256, 0, stream>>>(h4n, W5, y, out);
}
